// NewsEncoder_25374666785354
// MI455X (gfx1250) — matrix operand order repaired and checked by reading, not compiled, not run on hardware
//
#include <hip/hip_runtime.h>
#include <hip/hip_bf16.h>

#define NB   64     // B
#define NS   32     // S
#define NE   300    // E
#define NH   16     // H
#define ND   16     // D
#define NM   256    // M = H*D
#define NAH  200    // AH
#define ROWS (NB*NS)   // 2048
#define NSD  (NS*ND)   // 512
#define KE   320       // E padded to mult of 32

typedef __bf16 bf16_t;
typedef bf16_t   v16bf __attribute__((ext_vector_type(16)));
typedef float    v8f   __attribute__((ext_vector_type(8)));
typedef unsigned v4u   __attribute__((ext_vector_type(4)));

union frag16 { v4u q[2]; unsigned u[8]; v16bf v; };
union bfpair { unsigned u; struct { bf16_t lo, hi; } s; };

__device__ inline float fast_tanh(float x) {
#if __has_builtin(__builtin_amdgcn_tanhf)
    return __builtin_amdgcn_tanhf(x);
#else
    return tanhf(x);
#endif
}

// ---------------------------------------------------------------------------
// Pack f32 weight (K x 256 row-major) into fragment-swizzled bf16 pairs:
//   Bswz[(((kt*16 + nt)*32 + lane)*8 + r] = dword{ W[2*k2][n], W[2*k2+1][n] }
//   with k2 = kt*16 + (lane>>4)*8 + r,  n = nt*16 + (lane&15).
// Each lane's 8 fragment dwords are contiguous -> 2 x b128 loads in the GEMM.
// ---------------------------------------------------------------------------
__global__ void k_pack_w(const float* __restrict__ W, unsigned* __restrict__ pk,
                         int K, int Kpad) {
    int id = blockIdx.x * blockDim.x + threadIdx.x;
    if (id >= (Kpad >> 1) * NM) return;
    int r    = id & 7;
    int lane = (id >> 3) & 31;
    int nt   = (id >> 8) & 15;
    int kt   = id >> 12;
    int k2 = kt * 16 + (lane >> 4) * 8 + r;
    int n  = nt * 16 + (lane & 15);
    int k0 = 2 * k2;
    float f0 = (k0     < K) ? W[(size_t)k0       * NM + n] : 0.0f;
    float f1 = (k0 + 1 < K) ? W[(size_t)(k0 + 1) * NM + n] : 0.0f;
    bfpair p; p.s.lo = (bf16_t)f0; p.s.hi = (bf16_t)f1;
    pk[id] = p.u;
}

// ---------------------------------------------------------------------------
// Gather embedding rows, convert to bf16 pairs, zero-pad K 300 -> 320.
// Row-major pairs: apk[row*(KE/2) + k2].
// ---------------------------------------------------------------------------
__global__ void k_gather(const int* __restrict__ x, const float* __restrict__ emb,
                         unsigned* __restrict__ apk) {
    int id = blockIdx.x * blockDim.x + threadIdx.x;
    if (id >= ROWS * (KE >> 1)) return;
    int row = id / (KE >> 1), k2 = id % (KE >> 1);
    const float* er = emb + (size_t)x[row] * NE;
    int k0 = 2 * k2;
    float f0 = (k0     < NE) ? er[k0]     : 0.0f;
    float f1 = (k0 + 1 < NE) ? er[k0 + 1] : 0.0f;
    bfpair p; p.s.lo = (bf16_t)f0; p.s.hi = (bf16_t)f1;
    apk[id] = p.u;
}

// ---------------------------------------------------------------------------
// WMMA GEMM: C(ROWS x 256) = A(ROWS x KPAD, bf16 row-major) @ Bswz + bias
// grid (ROWS/16), block 128: 4 waves, each wave owns a 16x64 output strip.
// Per k-tile per wave: 2 A b128 loads + 8 B b128 loads + 4 wmma.
// KPAD and output type are compile-time.
// ---------------------------------------------------------------------------
template<int KPAD, bool OUT_BF16>
__global__ void __launch_bounds__(128)
k_gemm(const bf16_t* __restrict__ A, const unsigned* __restrict__ Bswz,
       const float* __restrict__ bias, void* __restrict__ Cout) {
    const int lane = threadIdx.x & 31;
    const int wave = threadIdx.x >> 5;
    const int row0 = blockIdx.x * 16;
    const int m    = lane & 15;
    const int hi   = lane >> 4;
    const int n0w  = wave * 64;

    const unsigned* arow = (const unsigned*)(A + (size_t)(row0 + m) * KPAD);

    v8f c0 = {}, c1 = {}, c2 = {}, c3 = {};
    for (int kt = 0; kt < KPAD / 32; ++kt) {
        // A 16x32 fragment: two contiguous groups of 4 pair-dwords
        frag16 fa;
        fa.q[0] = *(const v4u*)(arow + kt * 16 + (hi << 3));
        fa.q[1] = *(const v4u*)(arow + kt * 16 + 4 + (hi << 3));

        // B fragments: contiguous 32B per lane in the swizzled layout
        const unsigned* bb = Bswz + ((size_t)(kt * 16 + wave * 4) * 32 + lane) * 8;
        frag16 fb0, fb1, fb2, fb3;
        fb0.q[0] = *(const v4u*)(bb);            fb0.q[1] = *(const v4u*)(bb + 4);
        fb1.q[0] = *(const v4u*)(bb + 256);      fb1.q[1] = *(const v4u*)(bb + 260);
        fb2.q[0] = *(const v4u*)(bb + 512);      fb2.q[1] = *(const v4u*)(bb + 516);
        fb3.q[0] = *(const v4u*)(bb + 768);      fb3.q[1] = *(const v4u*)(bb + 772);
        if (kt + 1 < KPAD / 32)
            __builtin_prefetch(bb + 16 * 32 * 8, 0, 1);   // next k-tile of B

        c0 = __builtin_amdgcn_wmma_f32_16x16x32_bf16(false, fa.v, false, fb0.v, (short)0, c0, false, false);
        c1 = __builtin_amdgcn_wmma_f32_16x16x32_bf16(false, fa.v, false, fb1.v, (short)0, c1, false, false);
        c2 = __builtin_amdgcn_wmma_f32_16x16x32_bf16(false, fa.v, false, fb2.v, (short)0, c2, false, false);
        c3 = __builtin_amdgcn_wmma_f32_16x16x32_bf16(false, fa.v, false, fb3.v, (short)0, c3, false, false);
    }

    v8f acc[4] = { c0, c1, c2, c3 };
#pragma unroll
    for (int t = 0; t < 4; ++t) {
        const int n = n0w + t * 16 + m;
        const float bb = bias[n];
#pragma unroll
        for (int r = 0; r < 8; ++r) {
            int mr = row0 + (hi << 3) + r;
            float val = acc[t][r] + bb;
            if (OUT_BF16) ((bf16_t*)Cout)[(size_t)mr * NM + n] = (bf16_t)val;
            else          ((float*) Cout)[(size_t)mr * NM + n] = val;
        }
    }
}

// ---------------------------------------------------------------------------
// Fused additive attention per batch element (grid 64, block 256).
// ---------------------------------------------------------------------------
__global__ void __launch_bounds__(256)
k_attn(const float* __restrict__ q, const float* __restrict__ k,
       const float* __restrict__ v, const float* __restrict__ W_attn,
       const float* __restrict__ b_attn, const float* __restrict__ attn_v,
       bf16_t* __restrict__ ctxbf) {
    __shared__ float aqs[NS][NAH];
    __shared__ float aks[NS][NAH];
    __shared__ float sc[NS][NS];
    __shared__ float avs[NAH];
    __shared__ float qs16[NS][ND];
    __shared__ float ks16[NS][ND];

    const int b   = blockIdx.x;
    const int tid = threadIdx.x;
    const int i_own = tid >> 3;
    const int cbase = (tid & 7) * 64;

    float acc[64];
#pragma unroll
    for (int c = 0; c < 64; ++c) acc[c] = 0.0f;
    for (int idx = tid; idx < NAH; idx += 256) avs[idx] = attn_v[idx];

    const float* W1 = W_attn;
    const float* W2 = W_attn + ND * NAH;

    for (int h = 0; h < NH; ++h) {
        for (int idx = tid; idx < NS * ND; idx += 256) {
            int i = idx >> 4, d = idx & 15;
            size_t off = (size_t)(b * NS + i) * NM + h * ND + d;
            qs16[i][d] = q[off];
            ks16[i][d] = k[off];
        }
        __syncthreads();
        for (int idx = tid; idx < NS * NAH; idx += 256) {
            int i = idx / NAH, e = idx % NAH;
            float sq = b_attn[e], sk = 0.0f;
#pragma unroll
            for (int d = 0; d < ND; ++d) {
                sq += qs16[i][d] * W1[d * NAH + e];
                sk += ks16[i][d] * W2[d * NAH + e];
            }
            aqs[i][e] = sq;
            aks[i][e] = sk;
        }
        __syncthreads();
        for (int p = tid; p < NS * NS; p += 256) {
            int i = p >> 5, j = p & 31;
            float s = 0.0f;
            for (int e = 0; e < NAH; ++e)
                s += avs[e] * fast_tanh(aqs[i][e] + aks[j][e]);
            sc[i][j] = s;
        }
        __syncthreads();
        if (tid < NS) {
            int i = tid;
            float mx = -1e30f;
            for (int j = 0; j < NS; ++j) mx = fmaxf(mx, sc[i][j]);
            float sum = 0.0f;
            for (int j = 0; j < NS; ++j) {
                float e = __expf(sc[i][j] - mx);
                sc[i][j] = e;
                sum += e;
            }
            float inv = 1.0f / sum;
            for (int j = 0; j < NS; ++j) sc[i][j] *= inv;
        }
        __syncthreads();
        float vreg[ND];
#pragma unroll
        for (int d = 0; d < ND; ++d)
            vreg[d] = v[(size_t)(b * NS + i_own) * NM + h * ND + d];
#pragma unroll
        for (int cc = 0; cc < 64; ++cc) {
            int c = cbase + cc;
            acc[cc] += sc[i_own][c >> 4] * vreg[c & 15];
        }
        __syncthreads();
    }
#pragma unroll
    for (int cc = 0; cc < 64; ++cc)
        ctxbf[(size_t)(b * NS + i_own) * NSD + cbase + cc] = (bf16_t)acc[cc];
}

// ---------------------------------------------------------------------------
__global__ void k_rowmax(const float* __restrict__ of, float* __restrict__ out) {
    int id = blockIdx.x * blockDim.x + threadIdx.x;
    if (id >= NB * NM) return;
    int b = id >> 8, n = id & 255;
    float m = -1e30f;
    for (int i = 0; i < NS; ++i)
        m = fmaxf(m, of[(size_t)(b * NS + i) * NM + n]);
    out[(size_t)b * NM + n] = m;
}

// ---------------------------------------------------------------------------
extern "C" void kernel_launch(void* const* d_in, const int* in_sizes, int n_in,
                              void* d_out, int out_size, void* d_ws, size_t ws_size,
                              hipStream_t stream) {
    const int*   x      = (const int*)  d_in[0];
    const float* emb    = (const float*)d_in[1];
    const float* W_proj = (const float*)d_in[2];
    const float* b_proj = (const float*)d_in[3];
    const float* Wq     = (const float*)d_in[4];
    const float* bq     = (const float*)d_in[5];
    const float* Wk     = (const float*)d_in[6];
    const float* bk     = (const float*)d_in[7];
    const float* Wv     = (const float*)d_in[8];
    const float* bv     = (const float*)d_in[9];
    const float* W_attn = (const float*)d_in[10];
    const float* b_attn = (const float*)d_in[11];
    const float* attn_v = (const float*)d_in[12];
    const float* W_fc   = (const float*)d_in[13];
    const float* b_fc   = (const float*)d_in[14];
    float* out = (float*)d_out;
    (void)in_sizes; (void)n_in; (void)out_size; (void)ws_size;

    char* ws = (char*)d_ws;
    size_t off = 0;
    unsigned* apk   = (unsigned*)(ws + off); off += (size_t)ROWS * (KE >> 1) * 4;
    bf16_t*   hbf   = (bf16_t*) (ws + off); off += (size_t)ROWS * NM * 2;
    float*    qb    = (float*)  (ws + off); off += (size_t)ROWS * NM * 4;
    float*    kb    = (float*)  (ws + off); off += (size_t)ROWS * NM * 4;
    float*    vb    = (float*)  (ws + off); off += (size_t)ROWS * NM * 4;
    bf16_t*   ctxbf = (bf16_t*) (ws + off); off += (size_t)ROWS * NSD * 2;
    float*    ofull = (float*)  (ws + off); off += (size_t)ROWS * NM * 4;
    unsigned* pkWp  = (unsigned*)(ws + off); off += (size_t)(KE  >> 1) * NM * 4;
    unsigned* pkWq  = (unsigned*)(ws + off); off += (size_t)(NM  >> 1) * NM * 4;
    unsigned* pkWk  = (unsigned*)(ws + off); off += (size_t)(NM  >> 1) * NM * 4;
    unsigned* pkWv  = (unsigned*)(ws + off); off += (size_t)(NM  >> 1) * NM * 4;
    unsigned* pkWfc = (unsigned*)(ws + off); off += (size_t)(NSD >> 1) * NM * 4;

    k_pack_w<<<((KE  >> 1) * NM + 255) / 256, 256, 0, stream>>>(W_proj, pkWp,  NE,  KE);
    k_pack_w<<<((NM  >> 1) * NM + 255) / 256, 256, 0, stream>>>(Wq,     pkWq,  NM,  NM);
    k_pack_w<<<((NM  >> 1) * NM + 255) / 256, 256, 0, stream>>>(Wk,     pkWk,  NM,  NM);
    k_pack_w<<<((NM  >> 1) * NM + 255) / 256, 256, 0, stream>>>(Wv,     pkWv,  NM,  NM);
    k_pack_w<<<((NSD >> 1) * NM + 255) / 256, 256, 0, stream>>>(W_fc,   pkWfc, NSD, NSD);
    k_gather<<<(ROWS * (KE >> 1) + 255) / 256, 256, 0, stream>>>(x, emb, apk);

    dim3 gg(ROWS / 16), bb(128);
    k_gemm<KE,  true ><<<gg, bb, 0, stream>>>((const bf16_t*)apk, pkWp,  b_proj, hbf);
    k_gemm<NM,  false><<<gg, bb, 0, stream>>>(hbf,   pkWq,  bq, qb);
    k_gemm<NM,  false><<<gg, bb, 0, stream>>>(hbf,   pkWk,  bk, kb);
    k_gemm<NM,  false><<<gg, bb, 0, stream>>>(hbf,   pkWv,  bv, vb);
    k_attn<<<NB, 256, 0, stream>>>(qb, kb, vb, W_attn, b_attn, attn_v, ctxbf);
    k_gemm<NSD, false><<<gg, bb, 0, stream>>>(ctxbf, pkWfc, b_fc, ofull);
    k_rowmax<<<(NB * NM + 255) / 256, 256, 0, stream>>>(ofull, out);
}
